// hcc_kl_3_52793738002458
// MI455X (gfx1250) — matrix operand order repaired and checked by reading, not compiled, not run on hardware
//
#include <hip/hip_runtime.h>
#include <hip/hip_bf16.h>
#include <math.h>

// ---- problem constants (match reference) ----
#define PN   64          // identities
#define KC   8           // instances per (modality, identity)
#define CC   512         // softmax dim
#define NN   (3 * PN * KC)   // 1536 rows
#define MM   (3 * PN)        // 192 hard clusters
#define MID_N ((NN / 3) * 2) // 1024
#define MID_M ((MM / 3) * 2) // 128
#define MARGIN 6.0f
#define EPSV  1e-9f

typedef __bf16 bf16;
typedef __attribute__((ext_vector_type(16))) __bf16 v16bf;
typedef __attribute__((ext_vector_type(8)))  __bf16 v8bf;
typedef __attribute__((ext_vector_type(8)))  float  v8f;

// ---------------- Stage 0: zero the 4 global accumulators ----------------
__global__ void hcc_init_kernel(float* accum) {
    if (threadIdx.x < 4) accum[threadIdx.x] = 0.0f;
}

// ---------------- Stage 1: softmax + cluster centers ----------------
// grid = MM blocks (one hard cluster each), 256 threads = 8 waves (one row each)
__global__ void hcc_prep_kernel(const float* __restrict__ x,
                                const int*   __restrict__ pids,
                                bf16* __restrict__ xs_bf,
                                bf16* __restrict__ lx_bf,
                                bf16* __restrict__ hc_bf,
                                bf16* __restrict__ lh_bf,
                                float* __restrict__ ent_x,
                                float* __restrict__ ent_h,
                                int*   __restrict__ pidhc) {
    __shared__ float s_xs[KC][CC];     // 8 rows * 512 f32 = 16 KB
    __shared__ float s_red[256];

    const int cluster = blockIdx.x;            // 0..191
    const int wave    = threadIdx.x >> 5;      // 0..7 (row within cluster)
    const int lane    = threadIdx.x & 31;
    const int row     = cluster * KC + wave;   // 0..1535

    const float* xr = x + (size_t)row * CC;

    // wave32 softmax over 512 columns (16 per lane, coalesced stride-32)
    float v[16];
    float mx = -INFINITY;
#pragma unroll
    for (int t = 0; t < 16; ++t) { v[t] = xr[lane + 32 * t]; mx = fmaxf(mx, v[t]); }
#pragma unroll
    for (int o = 16; o >= 1; o >>= 1) mx = fmaxf(mx, __shfl_xor(mx, o, 32));
    float se = 0.0f;
#pragma unroll
    for (int t = 0; t < 16; ++t) { v[t] = __expf(v[t] - mx); se += v[t]; }
#pragma unroll
    for (int o = 16; o >= 1; o >>= 1) se += __shfl_xor(se, o, 32);
    const float inv = 1.0f / se;

    float ent = 0.0f;
#pragma unroll
    for (int t = 0; t < 16; ++t) {
        const int   c   = lane + 32 * t;
        const float xsv = v[t] * inv;
        const float lx  = __logf(fmaxf(xsv, EPSV));  // log(clip(xs, EPS))
        ent += xsv * lx;
        xs_bf[(size_t)row * CC + c] = (bf16)xsv;
        lx_bf[(size_t)row * CC + c] = (bf16)lx;
        s_xs[wave][c] = xsv;
    }
#pragma unroll
    for (int o = 16; o >= 1; o >>= 1) ent += __shfl_xor(ent, o, 32);
    if (lane == 0) ent_x[row] = ent;

    __syncthreads();

    // cluster center: mean of 8 rows; 256 threads * 2 columns
    float eh = 0.0f;
#pragma unroll
    for (int t = 0; t < 2; ++t) {
        const int c = threadIdx.x + 256 * t;
        float s = 0.0f;
#pragma unroll
        for (int rr = 0; rr < KC; ++rr) s += s_xs[rr][c];
        const float hc = s * (1.0f / KC);
        const float lh = __logf(fmaxf(hc, EPSV));
        eh += hc * lh;
        hc_bf[(size_t)cluster * CC + c] = (bf16)hc;
        lh_bf[(size_t)cluster * CC + c] = (bf16)lh;
    }
    s_red[threadIdx.x] = eh;
    __syncthreads();
#pragma unroll
    for (int s = 128; s >= 1; s >>= 1) {
        if (threadIdx.x < s) s_red[threadIdx.x] += s_red[threadIdx.x + s];
        __syncthreads();
    }
    if (threadIdx.x == 0) {
        ent_h[cluster] = s_red[0];
        pidhc[cluster] = pids[cluster * KC];
    }
}

// ---------------- Stage 2: dual bf16 WMMA GEMM + masked reductions ----------------
// One wave per 16x16 output tile. rowTiles = 96, colTiles = 12 -> 1152 waves.
// 128-thread blocks (4 waves) -> 288 blocks. Uniform control flow (EXEC all-1s).
__global__ void hcc_dist_kernel(const bf16* __restrict__ xs_bf,
                                const bf16* __restrict__ lx_bf,
                                const bf16* __restrict__ hc_bf,
                                const bf16* __restrict__ lh_bf,
                                const float* __restrict__ ent_x,
                                const float* __restrict__ ent_h,
                                const int*   __restrict__ pidhc,
                                const int*   __restrict__ pids,
                                float* __restrict__ accum) {
    const int wave = threadIdx.x >> 5;
    const int lane = threadIdx.x & 31;
    const int tile = blockIdx.x * 4 + wave;      // 0..1151
    const int rt = tile / (MM / 16);             // row tile 0..95
    const int ct = tile % (MM / 16);             // col tile 0..11

    const int half = lane & 15;                  // M for A, N for B/C/D
    const int sel  = lane >> 4;                  // K-half selector

    const int arow = rt * 16 + half;             // global row i of A-tile lane
    const int bcol = ct * 16 + half;             // global col j of B-tile lane

    // A layout (16-bit, 16x32): lanes 0-15 hold M=lane, K = {2v,2v+1} in v=0..3
    // (elements 0..7 -> K 0..7) and elements 8..15 -> K 16..23; lanes 16-31: K+8.
    const bf16* aL = lx_bf + (size_t)arow * CC + sel * 8;
    const bf16* aX = xs_bf + (size_t)arow * CC + sel * 8;
    // B layout: same K order as A: elements 0..7 -> K 0..7, 8..15 -> K 16..23; lanes 16-31: K+8.
    const bf16* bH = hc_bf + (size_t)bcol * CC + sel * 8;
    const bf16* bL = lh_bf + (size_t)bcol * CC + sel * 8;

    v8f acc1 = {};   // logx @ hcen^T
    v8f acc2 = {};   // xs   @ logh^T

#pragma unroll 4
    for (int k = 0; k < CC; k += 32) {
        v8bf a0 = *(const v8bf*)(aL + k);
        v8bf a1 = *(const v8bf*)(aL + k + 16);
        v16bf A1;
#pragma unroll
        for (int i = 0; i < 8; ++i) { A1[i] = a0[i]; A1[i + 8] = a1[i]; }

        v8bf c0 = *(const v8bf*)(aX + k);
        v8bf c1 = *(const v8bf*)(aX + k + 16);
        v16bf A2;
#pragma unroll
        for (int i = 0; i < 8; ++i) { A2[i] = c0[i]; A2[i + 8] = c1[i]; }

        v8bf h0 = *(const v8bf*)(bH + k);
        v8bf h1 = *(const v8bf*)(bH + k + 16);
        v8bf l0 = *(const v8bf*)(bL + k);
        v8bf l1 = *(const v8bf*)(bL + k + 16);
        v16bf B1, B2;
#pragma unroll
        for (int i = 0; i < 8; ++i) { B1[i] = h0[i]; B1[i + 8] = h1[i]; B2[i] = l0[i]; B2[i + 8] = l1[i]; }

        acc1 = __builtin_amdgcn_wmma_f32_16x16x32_bf16(
            false, A1, false, B1, (short)0, acc1, false, false);
        acc2 = __builtin_amdgcn_wmma_f32_16x16x32_bf16(
            false, A2, false, B2, (short)0, acc2, false, false);
    }

    // Epilogue: C/D layout -> VGPR r holds M = r + 8*sel, N = lane&15
    const int j   = ct * 16 + half;
    const int pj  = pidhc[j];
    const float ehj = ent_h[j];

    float psum = 0.0f, pcnt = 0.0f, nsum = 0.0f, ncnt = 0.0f;
#pragma unroll
    for (int r = 0; r < 8; ++r) {
        const int   m = r + sel * 8;
        const int   i = rt * 16 + m;
        const float d = ehj + ent_x[i] - acc1[r] - acc2[r];
        const bool  mk = (pids[i] == pj);
        const bool  region = ((i < MID_N) && (j >= MID_M)) ||
                             ((i >= MID_N) && (j < MID_M));
        if (region && mk) { psum += d; pcnt += 1.0f; }
        if (!mk)          { nsum += fmaxf(MARGIN - d, 0.0f); ncnt += 1.0f; }
    }
#pragma unroll
    for (int o = 16; o >= 1; o >>= 1) {
        psum += __shfl_xor(psum, o, 32);
        pcnt += __shfl_xor(pcnt, o, 32);
        nsum += __shfl_xor(nsum, o, 32);
        ncnt += __shfl_xor(ncnt, o, 32);
    }
    if (lane == 0) {
        atomicAdd(&accum[0], psum);
        atomicAdd(&accum[1], pcnt);
        atomicAdd(&accum[2], nsum);
        atomicAdd(&accum[3], ncnt);
    }
}

// ---------------- Stage 3: finalize ----------------
__global__ void hcc_final_kernel(const float* __restrict__ accum, float* out) {
    const float pc = fmaxf(accum[1], 1.0f);
    const float nc = fmaxf(accum[3], 1.0f);
    out[0] = accum[0] / pc + accum[2] / nc;   // K1 = K2 = 1
}

extern "C" void kernel_launch(void* const* d_in, const int* in_sizes, int n_in,
                              void* d_out, int out_size, void* d_ws, size_t ws_size,
                              hipStream_t stream) {
    const float* x    = (const float*)d_in[0];
    const int*   pids = (const int*)d_in[1];
    // d_in[2] = num_ids (device scalar) — dims are fixed by the reference shapes.

    // workspace layout (256B-aligned segments)
    char* ws = (char*)d_ws;
    size_t off = 0;
    auto take = [&](size_t bytes) { char* p = ws + off; off = (off + bytes + 255) & ~(size_t)255; return p; };
    bf16*  xs_bf = (bf16*)take((size_t)NN * CC * sizeof(bf16));
    bf16*  lx_bf = (bf16*)take((size_t)NN * CC * sizeof(bf16));
    bf16*  hc_bf = (bf16*)take((size_t)MM * CC * sizeof(bf16));
    bf16*  lh_bf = (bf16*)take((size_t)MM * CC * sizeof(bf16));
    float* ent_x = (float*)take((size_t)NN * sizeof(float));
    float* ent_h = (float*)take((size_t)MM * sizeof(float));
    int*   pidhc = (int*)take((size_t)MM * sizeof(int));
    float* accum = (float*)take(4 * sizeof(float));

    hcc_init_kernel<<<1, 32, 0, stream>>>(accum);
    hcc_prep_kernel<<<MM, 256, 0, stream>>>(x, pids, xs_bf, lx_bf, hc_bf, lh_bf,
                                            ent_x, ent_h, pidhc);
    const int numTiles = (NN / 16) * (MM / 16);   // 96 * 12 = 1152 waves
    hcc_dist_kernel<<<numTiles / 4, 128, 0, stream>>>(xs_bf, lx_bf, hc_bf, lh_bf,
                                                      ent_x, ent_h, pidhc, pids, accum);
    hcc_final_kernel<<<1, 1, 0, stream>>>(accum, (float*)d_out);
}
